// Attention_90546500534785
// MI455X (gfx1250) — matrix operand order repaired and checked by reading, not compiled, not run on hardware
//
#include <hip/hip_runtime.h>

// ---------------------------------------------------------------------------
// Self-attention (b=2, n=2048, dm=1024, 16 heads x 64) for MI455X / gfx1250.
// All matmuls via v_wmma_f32_16x16x32_bf16 (wave32). Compute-bound (~69 GFLOP,
// ~50MB traffic -> L2-resident on 192MB L2). GEMM main loop is a manual
// 2-stage ping-pong pipeline (#pragma unroll 1) so loads of the next k-step
// overlap the current WMMAs without register spills or rotation copies.
// ---------------------------------------------------------------------------

#define DMODEL 1024
#define INNERD 1024
#define NSEQ   2048
#define NBATCH 2
#define MTOT   (NBATCH * NSEQ)   // 4096 rows
#define NHEAD  16
#define DHEAD  64

typedef __attribute__((ext_vector_type(16))) __bf16 v16bf;
typedef __attribute__((ext_vector_type(8)))  float  v8f;

union FragU { v16bf v; uint4 q[2]; };

// A-layout fragment (16x32 bf16): lane<16 -> K = 0..15, lane>=16 ->
// K = 16..31. Caller folds the +16 element offset (kbA) into `p`.
__device__ __forceinline__ v16bf load_fragA(const __bf16* p) {
  FragU f;
  f.q[0] = *(const uint4*)(p);
  f.q[1] = *(const uint4*)(p + 8);
  return f.v;
}
// B-layout fragment (32x16 bf16): lane<16 -> K = 0..15, lane>=16 -> K = 16..31,
// col = lane%16. Caller folds the +16 element offset (kbB) into `p`; the 16
// bf16 are contiguous (two b128 loads).
__device__ __forceinline__ v16bf load_fragB(const __bf16* p) {
  FragU f;
  const uint4* q = (const uint4*)p;
  f.q[0] = q[0];
  f.q[1] = q[1];
  return f.v;
}

__device__ __forceinline__ v8f wmma_bf16(v16bf a, v16bf b, v8f c) {
  return __builtin_amdgcn_wmma_f32_16x16x32_bf16(false, a, false, b,
                                                 (short)0, c, false, false);
}

// ---------------------------------------------------------------------------
// 1) fp32 -> bf16 activation convert (4 elems/thread)
// ---------------------------------------------------------------------------
__global__ void k_f32_to_bf16(const float* __restrict__ x,
                              __bf16* __restrict__ y, int n) {
  int i = (blockIdx.x * blockDim.x + threadIdx.x) * 4;
  if (i + 3 < n) {
    float4 f = *(const float4*)(x + i);
    y[i + 0] = (__bf16)f.x;
    y[i + 1] = (__bf16)f.y;
    y[i + 2] = (__bf16)f.z;
    y[i + 3] = (__bf16)f.w;
  }
}

// ---------------------------------------------------------------------------
// 2) Pre-pack weight W[K,N] (row-major f32) into WMMA B-fragment order:
//    WP[((kt*ntiles + nt)*32 + lane)*16 + e] = W[kt*32 + kbB(lane) + e][nt*16 + lane%16]
//    so GEMM B loads are contiguous 32B per lane.
// ---------------------------------------------------------------------------
__global__ void k_pack_w(const float* __restrict__ W, __bf16* __restrict__ WP,
                         int K, int N) {
  int idx  = blockIdx.x * blockDim.x + threadIdx.x;
  int e    = idx & 15;
  int lane = (idx >> 4) & 31;
  int frag = idx >> 9;
  int ntiles = N >> 4;
  int nt = frag % ntiles;
  int kt = frag / ntiles;
  int k   = (kt << 5) + (lane & 16) + e;     // kbB = 0 or 16
  int col = (nt << 4) + (lane & 15);
  (void)K;
  WP[idx] = (__bf16)W[(size_t)k * N + col];
}

// ---------------------------------------------------------------------------
// Shared GEMM main loop: one wave computes a 16(M) x 64(N) f32 tile.
// Manual 2-stage ping-pong: compute on buffer0 while buffer1's 10 b128 loads
// are in flight, then swap. Producers write directly into the loop-carried
// fragment registers (no v_mov rotation, no compiler re-unroll, no spills).
// ---------------------------------------------------------------------------
__device__ __forceinline__ void gemm16x64(const __bf16* __restrict__ X,
                                          const __bf16* __restrict__ WP,
                                          int K, int ntiles, int mBase, int nt0,
                                          int lane, v8f acc[4]) {
  const int l16 = lane & 15;
  const int kbA = (lane & 16) ? 16 : 0;
  const __bf16* arow = X + (size_t)(mBase + l16) * K + kbA;
  const int kTiles = K >> 5;                 // 32 (even)
  const size_t wstride = (size_t)ntiles * 512;
  const __bf16* wb0 = WP + (size_t)nt0 * 512 + lane * 16;

  v16bf a0  = load_fragA(arow);
  v16bf b00 = load_fragB(wb0);
  v16bf b01 = load_fragB(wb0 + 512);
  v16bf b02 = load_fragB(wb0 + 1024);
  v16bf b03 = load_fragB(wb0 + 1536);

#pragma unroll 1
  for (int kt = 0; kt < kTiles; kt += 2) {
    const int kt1 = (kt + 1 < kTiles) ? kt + 1 : kt;   // clamp: no OOB addr
    const __bf16* a1p = arow + (kt1 << 5);
    const __bf16* w1p = wb0 + (size_t)kt1 * wstride;
    v16bf a1  = load_fragA(a1p);
    v16bf b10 = load_fragB(w1p);
    v16bf b11 = load_fragB(w1p + 512);
    v16bf b12 = load_fragB(w1p + 1024);
    v16bf b13 = load_fragB(w1p + 1536);
    acc[0] = wmma_bf16(a0, b00, acc[0]);
    acc[1] = wmma_bf16(a0, b01, acc[1]);
    acc[2] = wmma_bf16(a0, b02, acc[2]);
    acc[3] = wmma_bf16(a0, b03, acc[3]);

    const int kt2 = (kt + 2 < kTiles) ? kt + 2 : kt1;  // clamp (last iter)
    const __bf16* a2p = arow + (kt2 << 5);
    const __bf16* w2p = wb0 + (size_t)kt2 * wstride;
    a0  = load_fragA(a2p);
    b00 = load_fragB(w2p);
    b01 = load_fragB(w2p + 512);
    b02 = load_fragB(w2p + 1024);
    b03 = load_fragB(w2p + 1536);
    acc[0] = wmma_bf16(a1, b10, acc[0]);
    acc[1] = wmma_bf16(a1, b11, acc[1]);
    acc[2] = wmma_bf16(a1, b12, acc[2]);
    acc[3] = wmma_bf16(a1, b13, acc[3]);
  }
}

// ---------------------------------------------------------------------------
// 3) Q projection: Qb = (X @ Wq) * (1/sqrt(64))  -> bf16 row-major [4096,1024]
//    (softmax scale folded in here so the attention loop skips it)
// ---------------------------------------------------------------------------
__global__ void k_gemm_q(const __bf16* __restrict__ X,
                         const __bf16* __restrict__ WP,
                         __bf16* __restrict__ Q) {
  const int lane = threadIdx.x & 31;
  const int wave = threadIdx.x >> 5;
  const int l16  = lane & 15;
  const int kbC  = (lane & 16) ? 8 : 0;   // C-frag row offset
  const int mBase = blockIdx.y << 4;
  const int nt0   = ((blockIdx.x << 2) + wave) << 2;
  v8f acc[4] = {};
  gemm16x64(X, WP, DMODEL, INNERD >> 4, mBase, nt0, lane, acc);
  for (int t = 0; t < 4; ++t) {
    int col = ((nt0 + t) << 4) + l16;
    for (int r = 0; r < 8; ++r)
      Q[(size_t)(mBase + r + kbC) * INNERD + col] = (__bf16)(acc[t][r] * 0.125f);
  }
}

// ---------------------------------------------------------------------------
// 4) KV projection: K -> bf16 row-major [b,n,h,d]; V -> bf16 TRANSPOSED
//    [b,h,d,n] so attention P.V B-fragments read contiguous keys.
// ---------------------------------------------------------------------------
__global__ void k_gemm_kv(const __bf16* __restrict__ X,
                          const __bf16* __restrict__ WP,
                          __bf16* __restrict__ Kc, __bf16* __restrict__ Vt) {
  const int lane = threadIdx.x & 31;
  const int wave = threadIdx.x >> 5;
  const int l16  = lane & 15;
  const int kbC  = (lane & 16) ? 8 : 0;
  const int mBase = blockIdx.y << 4;
  const int nt0   = ((blockIdx.x << 2) + wave) << 2;
  v8f acc[4] = {};
  gemm16x64(X, WP, DMODEL, (2 * INNERD) >> 4, mBase, nt0, lane, acc);
  const int bb   = mBase >> 11;            // batch (tiles never straddle)
  const int nloc = mBase & (NSEQ - 1);
  for (int t = 0; t < 4; ++t) {
    int colBase = (nt0 + t) << 4;
    if (colBase < INNERD) {                // K half
      int col = colBase + l16;
      for (int r = 0; r < 8; ++r)
        Kc[(size_t)(mBase + r + kbC) * INNERD + col] = (__bf16)acc[t][r];
    } else {                               // V half, store transposed
      int inner = colBase + l16 - INNERD;
      int h = inner >> 6, d = inner & 63;
      __bf16* vrow = Vt + ((size_t)(bb * NHEAD + h) * DHEAD + d) * NSEQ;
      for (int r = 0; r < 8; ++r)
        vrow[nloc + r + kbC] = (__bf16)acc[t][r];
    }
  }
}

// ---------------------------------------------------------------------------
// 5) Flash attention: one wave per 16-query tile per (b,h); streams 32 keys
//    per step: 4 wmma (S) + online softmax + 4 wmma (P.V). V-fragment loads
//    are hoisted above the softmax VALU block so they fly under the shuffles/
//    exp; next step's K/V lines get global_prefetch hints. P goes through a
//    per-wave LDS tile to convert C-layout f32 -> A-layout bf16 fragment.
// ---------------------------------------------------------------------------
__global__ void k_attn(const __bf16* __restrict__ Q,
                       const __bf16* __restrict__ Kc,
                       const __bf16* __restrict__ Vt,
                       __bf16* __restrict__ O) {
  const int lane = threadIdx.x & 31;
  const int wave = threadIdx.x >> 5;
  const int l16  = lane & 15;
  const int hi   = (lane >> 4) & 1;
  const int kbA  = hi << 4;    // A-frag K offset
  const int kbB  = hi << 4;    // B-frag K offset
  const int kbC  = hi << 3;    // C-frag row offset
  const int bh = blockIdx.y;
  const int b = bh >> 4, h = bh & 15;
  const int qBase = (((blockIdx.x << 2) + wave) << 4);

  __shared__ __bf16 plds[4][16][32];       // 1KB per wave: P staging tile
  __bf16 (*pw)[32] = plds[wave];

  const __bf16* qrow = Q + (size_t)(b * NSEQ + qBase + l16) * INNERD
                         + h * DHEAD + kbA;
  const v16bf aq0 = load_fragA(qrow);        // d = 0..31
  const v16bf aq1 = load_fragA(qrow + 32);   // d = 32..63

  v8f o0 = {}, o1 = {}, o2 = {}, o3 = {};
  float m[8], l[8];
  for (int r = 0; r < 8; ++r) { m[r] = -3.0e38f; l[r] = 0.f; }

  const __bf16* kbase = Kc + (size_t)b * NSEQ * INNERD + h * DHEAD + kbB;
  const __bf16* vbase = Vt + ((size_t)bh * DHEAD + l16) * NSEQ + kbB;

#pragma unroll 1
  for (int j = 0; j < NSEQ; j += 32) {
    // ---- S = Qs . K^T for 32 keys (two 16-key column tiles) ----
    v8f s0 = {}, s1 = {};
    const __bf16* kr0 = kbase + (size_t)(j + l16) * INNERD;
    const __bf16* kr1 = kr0 + (size_t)16 * INNERD;
    s0 = wmma_bf16(aq0, load_fragB(kr0),      s0);
    s0 = wmma_bf16(aq1, load_fragB(kr0 + 32), s0);
    s1 = wmma_bf16(aq0, load_fragB(kr1),      s1);
    s1 = wmma_bf16(aq1, load_fragB(kr1 + 32), s1);

    // ---- V fragments issued early: latency hides under softmax VALU ----
    const __bf16* vr = vbase + j;
    v16bf bv0 = load_fragB(vr);
    v16bf bv1 = load_fragB(vr + (size_t)16 * NSEQ);
    v16bf bv2 = load_fragB(vr + (size_t)32 * NSEQ);
    v16bf bv3 = load_fragB(vr + (size_t)48 * NSEQ);

    // hint next j-step's K/V lines toward the caches
    __builtin_prefetch((const void*)(kr0 + (size_t)32 * INNERD), 0, 3);
    __builtin_prefetch((const void*)(kr1 + (size_t)32 * INNERD), 0, 3);
    __builtin_prefetch((const void*)(vr + 32), 0, 3);

    // ---- online softmax (rows live across 16-lane halves) ----
    for (int r = 0; r < 8; ++r) {
      float x0 = s0[r], x1 = s1[r];
      float mx = fmaxf(x0, x1);
      mx = fmaxf(mx, __shfl_xor(mx, 1, 32));
      mx = fmaxf(mx, __shfl_xor(mx, 2, 32));
      mx = fmaxf(mx, __shfl_xor(mx, 4, 32));
      mx = fmaxf(mx, __shfl_xor(mx, 8, 32));
      float mn = fmaxf(m[r], mx);
      float al = __expf(m[r] - mn);
      m[r] = mn;
      float p0 = __expf(x0 - mn);
      float p1 = __expf(x1 - mn);
      float sm = p0 + p1;
      sm += __shfl_xor(sm, 1, 32);
      sm += __shfl_xor(sm, 2, 32);
      sm += __shfl_xor(sm, 4, 32);
      sm += __shfl_xor(sm, 8, 32);
      l[r] = l[r] * al + sm;
      pw[r + kbC][l16]      = (__bf16)p0;   // C layout -> row-major LDS
      pw[r + kbC][l16 + 16] = (__bf16)p1;
      o0[r] *= al; o1[r] *= al; o2[r] *= al; o3[r] *= al;
    }
    asm volatile("s_wait_dscnt 0" ::: "memory");  // P stores visible to reads

    // ---- O += P . V (V pre-transposed: contiguous keys per d-row) ----
    v16bf pa = load_fragA(&pw[l16][kbA]);
    o0 = wmma_bf16(pa, bv0, o0);
    o1 = wmma_bf16(pa, bv1, o1);
    o2 = wmma_bf16(pa, bv2, o2);
    o3 = wmma_bf16(pa, bv3, o3);
  }

  // ---- normalize and store O as bf16 [b, n, h*d] ----
  __bf16* orow = O + (size_t)(b * NSEQ + qBase) * INNERD + h * DHEAD;
  for (int r = 0; r < 8; ++r) {
    float inv = 1.0f / l[r];
    __bf16* p = orow + (size_t)(r + kbC) * INNERD + l16;
    p[0]  = (__bf16)(o0[r] * inv);
    p[16] = (__bf16)(o1[r] * inv);
    p[32] = (__bf16)(o2[r] * inv);
    p[48] = (__bf16)(o3[r] * inv);
  }
}

// ---------------------------------------------------------------------------
// 6) Output projection: out = Ob @ Wo + bo  (f32 output)
// ---------------------------------------------------------------------------
__global__ void k_gemm_o(const __bf16* __restrict__ X,
                         const __bf16* __restrict__ WP,
                         const float* __restrict__ bo,
                         float* __restrict__ out) {
  const int lane = threadIdx.x & 31;
  const int wave = threadIdx.x >> 5;
  const int l16  = lane & 15;
  const int kbC  = (lane & 16) ? 8 : 0;
  const int mBase = blockIdx.y << 4;
  const int nt0   = ((blockIdx.x << 2) + wave) << 2;
  v8f acc[4] = {};
  gemm16x64(X, WP, INNERD, DMODEL >> 4, mBase, nt0, lane, acc);
  for (int t = 0; t < 4; ++t) {
    int col = ((nt0 + t) << 4) + l16;
    float bias = bo[col];
    for (int r = 0; r < 8; ++r)
      out[(size_t)(mBase + r + kbC) * DMODEL + col] = acc[t][r] + bias;
  }
}

// ---------------------------------------------------------------------------
extern "C" void kernel_launch(void* const* d_in, const int* in_sizes, int n_in,
                              void* d_out, int out_size, void* d_ws,
                              size_t ws_size, hipStream_t stream) {
  (void)in_sizes; (void)n_in; (void)out_size; (void)ws_size;
  const float* queries = (const float*)d_in[0];
  const float* Wq      = (const float*)d_in[1];
  const float* Wkv     = (const float*)d_in[2];
  const float* Wo      = (const float*)d_in[3];
  const float* bo      = (const float*)d_in[4];
  float* out = (float*)d_out;

  // Workspace carve (48 MB total)
  char* ws = (char*)d_ws;
  __bf16* Xb   = (__bf16*)(ws);                    //  8 MB activations bf16
  __bf16* WqP  = (__bf16*)(ws + ( 8ull << 20));    //  2 MB packed Wq
  __bf16* WkvP = (__bf16*)(ws + (10ull << 20));    //  4 MB packed Wkv
  __bf16* WoP  = (__bf16*)(ws + (14ull << 20));    //  2 MB packed Wo
  __bf16* Qb   = (__bf16*)(ws + (16ull << 20));    //  8 MB scaled Q
  __bf16* Kb   = (__bf16*)(ws + (24ull << 20));    //  8 MB K
  __bf16* Vt   = (__bf16*)(ws + (32ull << 20));    //  8 MB V transposed
  __bf16* Ob   = (__bf16*)(ws + (40ull << 20));    //  8 MB attn output

  const int nact = MTOT * DMODEL;                  // 4,194,304
  k_f32_to_bf16<<<nact / 1024, 256, 0, stream>>>(queries, Xb, nact);

  k_pack_w<<<(DMODEL * INNERD) / 256,     256, 0, stream>>>(Wq,  WqP,  DMODEL, INNERD);
  k_pack_w<<<(DMODEL * 2 * INNERD) / 256, 256, 0, stream>>>(Wkv, WkvP, DMODEL, 2 * INNERD);
  k_pack_w<<<(INNERD * DMODEL) / 256,     256, 0, stream>>>(Wo,  WoP,  INNERD, DMODEL);

  // 4 waves/block, each wave owns a 16x64 tile (grid.x covers N/256).
  k_gemm_q <<<dim3(INNERD / 256,     MTOT / 16), 128, 0, stream>>>(Xb, WqP, Qb);
  k_gemm_kv<<<dim3(2 * INNERD / 256, MTOT / 16), 128, 0, stream>>>(Xb, WkvP, Kb, Vt);

  // 4 waves/block, each wave = one 16-query tile; grid.y = b*h.
  k_attn<<<dim3(NSEQ / 64, NBATCH * NHEAD), 128, 0, stream>>>(Qb, Kb, Vt, Ob);

  k_gemm_o<<<dim3(DMODEL / 256, MTOT / 16), 128, 0, stream>>>(Ob, WoP, bo, out);
}
